// VQ_VAE_32323923870349
// MI455X (gfx1250) — compile-verified, matrix operand order checked by reading, not run on hardware
//
#include <hip/hip_runtime.h>
#include <hip/hip_bf16.h>

// ---------------------------------------------------------------------------
// VQ-VAE forward for MI455X (gfx1250, wave32): WMMA bf16 + TDM tile loads
// ---------------------------------------------------------------------------

typedef __attribute__((ext_vector_type(16))) __bf16 v16bf;
typedef __attribute__((ext_vector_type(8)))  float  v8f;
typedef __attribute__((ext_vector_type(4)))  unsigned int v4u;
typedef __attribute__((ext_vector_type(8)))  int    v8i;
typedef __attribute__((ext_vector_type(4)))  int    v4i;

#define BATCH   32768
#define N_EMB   512
#define EMB_DIM 64

__device__ __forceinline__ unsigned short f2bf(float f) {
  union { float f; unsigned int u; } c; c.f = f;
  unsigned int u = c.u;
  return (unsigned short)((u + 0x7FFFu + ((u >> 16) & 1u)) >> 16); // RNE
}

// ----------------------------- small kernels -------------------------------

__global__ void cvt_bf16_kernel(const float* __restrict__ src,
                                unsigned short* __restrict__ dst, long n) {
  long i = (long)blockIdx.x * blockDim.x + threadIdx.x;
  if (i < n) dst[i] = f2bf(src[i]);
}

__global__ void emb_norm_kernel(const float* __restrict__ emb,
                                float* __restrict__ e2) {
  int c = blockIdx.x * blockDim.x + threadIdx.x;
  if (c < N_EMB) {
    float s = 0.f;
    #pragma unroll 8
    for (int k = 0; k < EMB_DIM; ++k) { float v = emb[c * EMB_DIM + k]; s += v * v; }
    e2[c] = s;
  }
}

// --------------------- TDM descriptor + issue helper -----------------------
// Issues TENSOR_LOAD_TO_LDS of a tile_rows x tile_cols (elements, bf16) tile
// whose top-left element is `gptr` (row pitch = `pitch` elements). Hardware
// zero-fills where the dim counters exceed (rows_rem, cols_rem): we pass the
// REMAINING tensor extent measured from the tile origin, so OOB rows/cols of
// the tile land as zeros in LDS. LDS layout: packed tile, row stride =
// tile_cols elements (pad disabled).
__device__ __forceinline__ void tdm_load_tile_bf16(
    unsigned lds_off, const unsigned short* gptr, long pitch,
    unsigned rows_rem, unsigned cols_rem, unsigned tile_rows, unsigned tile_cols) {
  unsigned long long ga = (unsigned long long)(size_t)gptr;

  v4u g0;
  g0.x = 1u;                                      // count=1, user mode
  g0.y = lds_off;                                 // lds_addr (byte offset)
  g0.z = (unsigned)(ga & 0xFFFFFFFFu);            // global_addr[31:0]
  g0.w = (unsigned)((ga >> 32) & 0x01FFFFFFu)     // global_addr[56:32]
         | (2u << 30);                            // type=2 ("image")

  unsigned pitch32 = (unsigned)pitch;             // tensor_dim0_stride (elements)
  v8i g1;
  g1[0] = (int)(1u << 16);                        // wg_mask=0, data_size=1 (2B), pad off
  g1[1] = (int)((cols_rem & 0xFFFFu) << 16);      // tensor_dim0[15:0]  @ bits 63:48
  g1[2] = (int)((cols_rem >> 16) |                // tensor_dim0[31:16] @ bits 79:64
                ((rows_rem & 0xFFFFu) << 16));    // tensor_dim1[15:0]  @ bits 95:80
  g1[3] = (int)(((rows_rem >> 16) & 0xFFFFu) |    // tensor_dim1[31:16] @ bits 111:96
                (tile_cols << 16));               // tile_dim0          @ bits 127:112
  g1[4] = (int)(tile_rows & 0xFFFFu);             // tile_dim1; tile_dim2=0 (2D)
  g1[5] = (int)pitch32;                           // tensor_dim0_stride[31:0]
  g1[6] = 0;                                      // stride[47:32]=0, dim1_stride lo=0
  g1[7] = 0;

  v4i gz4 = {0, 0, 0, 0};
  v8i gz8 = {0, 0, 0, 0, 0, 0, 0, 0};
  // 6-arg form (clang-23 / therock-10.0 lane): (g0, g1, g2, g3, g4, cpol)
  __builtin_amdgcn_tensor_load_to_lds(g0, g1, gz4, gz4, gz8, 0);
}

// ----------------------------- WMMA GEMM -----------------------------------
// C[M,N] = act(A[M,K] @ W[N,K]^T + bias)
// act: 0 = none -> f32 out, 1 = relu -> bf16 out, 2 = sigmoid -> f32 out
// Block tile 128(M) x 128(N), K-chunk 64, double-buffered LDS fed by TDM.
// 8 waves as 2(M) x 4(N); wave tile 64x32 -> 16 WMMAs per chunk per wave.

union FragU { uint4 q[2]; v16bf v; };

__global__ __launch_bounds__(256)
void gemm_bf16_wmma(const unsigned short* __restrict__ A,
                    const unsigned short* __restrict__ W,
                    const float* __restrict__ bias,
                    unsigned short* __restrict__ outb,
                    float* __restrict__ outf,
                    int M, int N, int K, int act) {
  __shared__ unsigned short As[2][128][64];   // 16 KB per buffer
  __shared__ unsigned short Ws[2][128][64];   // 16 KB per buffer (64 KB total)

  const int tid  = threadIdx.x;
  const int wave = tid >> 5;
  const int lane = tid & 31;
  const int hl   = lane >> 4;   // lane half (0 / 1)
  const int l16  = lane & 15;
  const int wm   = wave >> 2;   // 0..1 : M slot (64 rows)
  const int wn   = wave & 3;    // 0..3 : N slot (32 cols)
  const int m0   = blockIdx.y * 128;
  const int n0   = blockIdx.x * 128;

  const v8f vz = {0.f, 0.f, 0.f, 0.f, 0.f, 0.f, 0.f, 0.f};
  v8f acc[4][2];
  #pragma unroll
  for (int i = 0; i < 4; ++i) { acc[i][0] = vz; acc[i][1] = vz; }

  const int nch = (K + 63) >> 6;   // number of 64-K chunks

  // Issue TDM loads for a chunk into LDS buffer `b` (wave 0 only).
  auto issue = [&](int t, int b) {
    int k0 = t << 6;
    tdm_load_tile_bf16((unsigned)(size_t)&As[b][0][0],
                       A + (size_t)m0 * K + k0, K,
                       (unsigned)(M - m0), (unsigned)(K - k0), 128u, 64u);
    tdm_load_tile_bf16((unsigned)(size_t)&Ws[b][0][0],
                       W + (size_t)n0 * K + k0, K,
                       (unsigned)(N > n0 ? N - n0 : 0), (unsigned)(K - k0), 128u, 64u);
  };

  if (wave == 0) issue(0, 0);

  for (int t = 0; t < nch; ++t) {
    const int pb = t & 1;
    if (wave == 0) {
      if (t + 1 < nch) {
        issue(t + 1, (t + 1) & 1);               // overlap DMA with compute
        __builtin_amdgcn_s_wait_tensorcnt(2);    // chunk t's 2 loads done (in-order)
      } else {
        __builtin_amdgcn_s_wait_tensorcnt(0);
      }
    }
    __syncthreads();

    // --- two 32-K WMMA substeps over the staged 64-K chunk ---
    // 16-bit fragment layout: lane<16: elems 0-7 = K0-7, 8-15 = K16-23;
    // lane>=16: K8-15, K24-31 (relative to substep origin ks).
    #pragma unroll
    for (int ks = 0; ks < 64; ks += 32) {
      FragU af[4], bfr[2];
      #pragma unroll
      for (int i = 0; i < 4; ++i) {
        int r = wm * 64 + i * 16 + l16;
        af[i].q[0] = *(const uint4*)&As[pb][r][ks + hl * 8];
        af[i].q[1] = *(const uint4*)&As[pb][r][ks + 16 + hl * 8];
      }
      #pragma unroll
      for (int j = 0; j < 2; ++j) {
        int r = wn * 32 + j * 16 + l16;
        bfr[j].q[0] = *(const uint4*)&Ws[pb][r][ks + hl * 8];
        bfr[j].q[1] = *(const uint4*)&Ws[pb][r][ks + 16 + hl * 8];
      }
      #pragma unroll
      for (int i = 0; i < 4; ++i)
        #pragma unroll
        for (int j = 0; j < 2; ++j)
          acc[i][j] = __builtin_amdgcn_wmma_f32_16x16x32_bf16(
              false, af[i].v, false, bfr[j].v, (short)0, acc[i][j], false, false);
    }
    __syncthreads();   // LDS buffer free before wave0 re-targets it
  }

  // --- epilogue: bias + activation. C/D layout: M = r + 8*hl, N = l16 ---
  #pragma unroll
  for (int j = 0; j < 2; ++j) {
    int n = n0 + wn * 32 + j * 16 + l16;
    if (n >= N) continue;
    float bb = bias[n];
    #pragma unroll
    for (int i = 0; i < 4; ++i) {
      int mbase = m0 + wm * 64 + i * 16 + hl * 8;
      #pragma unroll
      for (int r = 0; r < 8; ++r) {
        float v = acc[i][j][r] + bb;
        size_t idx = (size_t)(mbase + r) * N + n;
        if (act == 1) {
          v = v > 0.f ? v : 0.f;
          outb[idx] = f2bf(v);
        } else if (act == 2) {
          outf[idx] = 1.f / (1.f + __expf(-v));
        } else {
          outf[idx] = v;
        }
      }
    }
  }
}

// ----------------------------- VQ lookup -----------------------------------
// One wave (32 lanes) per sample; each lane owns dims {lane, lane+32}.

__global__ __launch_bounds__(256)
void vq_kernel(const float* __restrict__ z, const float* __restrict__ emb,
               const float* __restrict__ e2, unsigned short* __restrict__ qb,
               float* __restrict__ loss_accum) {
  int s = blockIdx.x * 8 + (threadIdx.x >> 5);
  int lane = threadIdx.x & 31;
  size_t zbase = (size_t)s * EMB_DIM;
  float z0 = z[zbase + lane];
  float z1 = z[zbase + 32 + lane];

  float zn = z0 * z0 + z1 * z1;
  for (int o = 16; o > 0; o >>= 1) zn += __shfl_xor(zn, o, 32);

  float best = 3.4e38f;
  int bidx = 0;
  for (int c = 0; c < N_EMB; ++c) {
    float d = z0 * emb[c * EMB_DIM + lane] + z1 * emb[c * EMB_DIM + 32 + lane];
    for (int o = 16; o > 0; o >>= 1) d += __shfl_xor(d, o, 32);
    float dist = zn + e2[c] - d;             // matches ref (no factor 2)
    if (dist < best) { best = dist; bidx = c; }
  }

  float q0 = emb[bidx * EMB_DIM + lane];
  float q1 = emb[bidx * EMB_DIM + 32 + lane];
  qb[zbase + lane]      = f2bf(q0);          // q_st == q in forward
  qb[zbase + 32 + lane] = f2bf(q1);

  float l = (q0 - z0) * (q0 - z0) + (q1 - z1) * (q1 - z1);
  for (int o = 16; o > 0; o >>= 1) l += __shfl_xor(l, o, 32);
  if (lane == 0) atomicAdd(loss_accum, l);
}

__global__ void finalize_loss(const float* __restrict__ accum,
                              float* __restrict__ out, int out_size) {
  // loss = code + 0.25*commit = 1.25 * mean((q-z)^2)
  out[out_size - 1] = accum[0] * (1.25f / ((float)BATCH * (float)EMB_DIM));
}

// ----------------------------- launcher ------------------------------------

extern "C" void kernel_launch(void* const* d_in, const int* in_sizes, int n_in,
                              void* d_out, int out_size, void* d_ws, size_t ws_size,
                              hipStream_t stream) {
  const float* x   = (const float*)d_in[0];
  const float* emb = (const float*)d_in[1];
  const float* We1 = (const float*)d_in[2];
  const float* be1 = (const float*)d_in[3];
  const float* We2 = (const float*)d_in[4];
  const float* be2 = (const float*)d_in[5];
  const float* We3 = (const float*)d_in[6];
  const float* be3 = (const float*)d_in[7];
  const float* Wd1 = (const float*)d_in[8];
  const float* bd1 = (const float*)d_in[9];
  const float* Wd2 = (const float*)d_in[10];
  const float* bd2 = (const float*)d_in[11];
  const float* Wd3 = (const float*)d_in[12];
  const float* bd3 = (const float*)d_in[13];
  float* out = (float*)d_out;

  char* ws = (char*)d_ws;
  size_t off = 0;
  auto take = [&](size_t bytes) -> char* {
    char* p = ws + off;
    off = (off + bytes + 255) & ~(size_t)255;
    return p;
  };

  unsigned short* xb   = (unsigned short*)take((size_t)BATCH * 784 * 2);
  unsigned short* buf1 = (unsigned short*)take((size_t)BATCH * 1024 * 2); // h1 / d1
  unsigned short* buf2 = (unsigned short*)take((size_t)BATCH * 1024 * 2); // h2 / d2
  float*          zf   = (float*)take((size_t)BATCH * EMB_DIM * 4);
  unsigned short* qb   = (unsigned short*)take((size_t)BATCH * EMB_DIM * 2);
  unsigned short* We1b = (unsigned short*)take((size_t)1024 * 784 * 2);
  unsigned short* We2b = (unsigned short*)take((size_t)1024 * 1024 * 2);
  unsigned short* We3b = (unsigned short*)take((size_t)64 * 1024 * 2);
  unsigned short* Wd1b = (unsigned short*)take((size_t)1024 * 64 * 2);
  unsigned short* Wd2b = (unsigned short*)take((size_t)1024 * 1024 * 2);
  unsigned short* Wd3b = (unsigned short*)take((size_t)784 * 1024 * 2);
  float*          e2   = (float*)take(N_EMB * 4);
  float*          acc  = (float*)take(4);

  auto cvt = [&](const float* s, unsigned short* d, long n) {
    cvt_bf16_kernel<<<dim3((unsigned)((n + 255) / 256)), 256, 0, stream>>>(s, d, n);
  };
  cvt(x,   xb,   (long)BATCH * 784);
  cvt(We1, We1b, 1024L * 784);
  cvt(We2, We2b, 1024L * 1024);
  cvt(We3, We3b, 64L * 1024);
  cvt(Wd1, Wd1b, 1024L * 64);
  cvt(Wd2, Wd2b, 1024L * 1024);
  cvt(Wd3, Wd3b, 784L * 1024);

  emb_norm_kernel<<<dim3(2), 256, 0, stream>>>(emb, e2);
  (void)hipMemsetAsync(acc, 0, sizeof(float), stream);

  const dim3 blk(256);
  auto gemm = [&](const unsigned short* Ab, const unsigned short* Wb,
                  const float* b, unsigned short* ob, float* of,
                  int M, int N, int K, int act) {
    dim3 grid((N + 127) / 128, M / 128);
    gemm_bf16_wmma<<<grid, blk, 0, stream>>>(Ab, Wb, b, ob, of, M, N, K, act);
  };

  // Encoder
  gemm(xb,   We1b, be1, buf1, nullptr, BATCH, 1024, 784,  1); // relu
  gemm(buf1, We2b, be2, buf2, nullptr, BATCH, 1024, 1024, 1); // relu
  gemm(buf2, We3b, be3, nullptr, zf,   BATCH, 64,   1024, 0); // z (f32)

  // VQ
  vq_kernel<<<dim3(BATCH / 8), 256, 0, stream>>>(zf, emb, e2, qb, acc);

  // Decoder (reuse buf1/buf2)
  gemm(qb,   Wd1b, bd1, buf1, nullptr, BATCH, 1024, 64,   1); // relu
  gemm(buf1, Wd2b, bd2, buf2, nullptr, BATCH, 1024, 1024, 1); // relu
  gemm(buf2, Wd3b, bd3, nullptr, out,  BATCH, 784,  1024, 2); // sigmoid -> d_out

  finalize_loss<<<dim3(1), 1, 0, stream>>>(acc, out, out_size);
}
